// MultiHeadSelfAttention_46686294507626
// MI455X (gfx1250) — compile-verified, matrix operand order checked by reading, not run on hardware
//
#include <hip/hip_runtime.h>
#include <hip/hip_bf16.h>

// ---------------------------------------------------------------------------
// MHA for MI455X (gfx1250, wave32).  All GEMM-shaped math runs on
// v_wmma_f32_16x16x32_bf16 with a bf16x3 (hi/lo) error-compensated split.
// hi/lo planes are pre-split ONCE into workspace; hot loops stage tiles with
// DOUBLE-BUFFERED async global->LDS B128 DMA (ASYNCcnt) overlapped with the
// WMMA pipe, and read fragments as aligned B128 pairs.
// ---------------------------------------------------------------------------

#define D_   1024
#define T_   2048
#define B_   2
#define H_   16
#define HD_  64
#define M_   (B_ * T_)          // 4096 token rows

typedef __bf16        v16bf __attribute__((ext_vector_type(16)));
typedef float         v8f   __attribute__((ext_vector_type(8)));
typedef unsigned int  v8u   __attribute__((ext_vector_type(8)));
typedef unsigned short u16;

// Fragment K-mapping (ISA 7.12.2, 16-bit 16x32 A/B):
//   lane<16 : K in {0..7} U {16..23};  lane>=16 : K in {8..15} U {24..31}
// Element pairs are K-consecutive and group into two aligned B128s per plane.
__device__ __forceinline__ v8u pack8(uint4 a, uint4 b) {
  v8u u;
  u[0] = a.x; u[1] = a.y; u[2] = a.z; u[3] = a.w;
  u[4] = b.x; u[5] = b.y; u[6] = b.z; u[7] = b.w;
  return u;
}

// One fragment plane from a (16B-aligned) row of u32 bf16-pairs.
// half = lane>>4 ; chW = 16*chunk words for 64-wide rows, else 0.
__device__ __forceinline__ v16bf frag_row(const unsigned int* row, int half, int chW) {
  const uint4* p4 = (const uint4*)(row + chW);
  return __builtin_bit_cast(v16bf, pack8(p4[half], p4[half + 2]));
}

// D = A*B + C with bf16x3 compensation: lo*hi + hi*lo + hi*hi
__device__ __forceinline__ v8f wmma3(v16bf ahi, v16bf alo, v16bf bhi, v16bf blo, v8f c) {
  c = __builtin_amdgcn_wmma_f32_16x16x32_bf16(false, alo, false, bhi, (short)0, c, false, false);
  c = __builtin_amdgcn_wmma_f32_16x16x32_bf16(false, ahi, false, blo, (short)0, c, false, false);
  c = __builtin_amdgcn_wmma_f32_16x16x32_bf16(false, ahi, false, bhi, (short)0, c, false, false);
  return c;
}

struct bf2 { u16 hi, lo; };
__device__ __forceinline__ bf2 split_bf16(float f) {
  __bf16 h = (__bf16)f;
  bf2 r;
  r.hi = __builtin_bit_cast(u16, h);
  r.lo = __builtin_bit_cast(u16, (__bf16)(f - (float)h));
  return r;
}

// Async global->LDS DMA (CDNA5 async path, tracked by ASYNCcnt).
// Generic LDS pointer low 32 bits == LDS byte address (ISA 10.2 aperture map).
__device__ __forceinline__ void async_b128(void* lds, const void* gaddr) {
  unsigned l = (unsigned)(unsigned long long)lds;
  unsigned long long g = (unsigned long long)gaddr;
  asm volatile("global_load_async_to_lds_b128 %0, %1, off"
               :: "v"(l), "v"(g) : "memory");
}
__device__ __forceinline__ void wait_async() {
  asm volatile("s_wait_asynccnt 0x0" ::: "memory");
}

// reduce across the 16 lanes of a wave32 half (xor masks < 16 stay in-half)
__device__ __forceinline__ float rmax16(float v) {
  #pragma unroll
  for (int m = 1; m < 16; m <<= 1) v = fmaxf(v, __shfl_xor(v, m, 32));
  return v;
}
__device__ __forceinline__ float rsum16(float v) {
  #pragma unroll
  for (int m = 1; m < 16; m <<= 1) v += __shfl_xor(v, m, 32);
  return v;
}

// ---------------------------------------------------------------------------
// Kernel 0: one-time fp32 -> (bf16 hi, bf16 lo) plane split
// ---------------------------------------------------------------------------
__global__ __launch_bounds__(256)
void split_kernel(const float* __restrict__ src, u16* __restrict__ hi,
                  u16* __restrict__ lo, int n) {
  int i = blockIdx.x * 256 + threadIdx.x;
  if (i < n) {
    bf2 s = split_bf16(src[i]);
    hi[i] = s.hi;
    lo[i] = s.lo;
  }
}

// ---------------------------------------------------------------------------
// GEMM tile body on pre-split planes: Y[64x32 block] = X[M,K] * W[N,K]^T.
// Double-buffered async staging; row stride 20 words (16B-aligned,
// conflict-free mod 64 banks).  8 waves -> one 16x16 C tile each.
// ---------------------------------------------------------------------------
__device__ __forceinline__ v8f gemm_tile_planes(
    const unsigned int* __restrict__ XH, const unsigned int* __restrict__ XL,
    const unsigned int* __restrict__ WH, const unsigned int* __restrict__ WL,
    int m0, int n0, int Kd2,
    unsigned int (*XsH)[64][20], unsigned int (*XsL)[64][20],
    unsigned int (*WsH)[32][20], unsigned int (*WsL)[32][20]) {
  const int tid = threadIdx.x, lane = tid & 31, wave = tid >> 5;
  const int wy = wave >> 1, wx = wave & 1, l15 = lane & 15, half = lane >> 4;
  const int ar = wy * 16 + l15;   // A: lane = row
  const int br = wx * 16 + l15;   // B: lane = output column (row of W)
  const int xr = tid >> 2, xc = (tid & 3) * 4;     // X stage: 1 uint4/thread
  const int wr = tid >> 2, wc = (tid & 3) * 4;     // W stage: tid<128 only

  auto stage = [&](int buf, int k0) {
    size_t gx = (size_t)(m0 + xr) * Kd2 + k0 + xc;
    async_b128(&XsH[buf][xr][xc], XH + gx);
    async_b128(&XsL[buf][xr][xc], XL + gx);
    if (tid < 128) {
      size_t gw = (size_t)(n0 + wr) * Kd2 + k0 + wc;
      async_b128(&WsH[buf][wr][wc], WH + gw);
      async_b128(&WsL[buf][wr][wc], WL + gw);
    }
  };

  v8f acc = {};
  stage(0, 0);                                     // prologue
  int buf = 0;
  for (int k0 = 0; k0 < Kd2; k0 += 16, buf ^= 1) { // 16 u32 = 32 bf16 of K
    wait_async();                                  // own tile-t DMAs done
    __syncthreads();                               // everyone's done
    if (k0 + 16 < Kd2) stage(buf ^ 1, k0 + 16);    // overlap next DMA w/ math

    v16bf ah = frag_row(&XsH[buf][ar][0], half, 0);
    v16bf al = frag_row(&XsL[buf][ar][0], half, 0);
    v16bf bh = frag_row(&WsH[buf][br][0], half, 0);
    v16bf bl = frag_row(&WsL[buf][br][0], half, 0);
    acc = wmma3(ah, al, bh, bl, acc);
  }
  return acc;
}

// ---------------------------------------------------------------------------
// Kernel 1: fused QKV projection -> pre-split Q/K/V planes.
//   Q,K: [bh][t][hd]   V: [bh][hd][t] (transposed for PV B-fragments)
//   Q gets the 1/8 softmax scale folded in (exact power of two).
// ---------------------------------------------------------------------------
__global__ __launch_bounds__(256)
void qkv_kernel(const unsigned int* __restrict__ XH, const unsigned int* __restrict__ XL,
                const unsigned int* __restrict__ WH, const unsigned int* __restrict__ WL,
                const float* __restrict__ b_qkv,
                u16* __restrict__ qh, u16* __restrict__ ql,
                u16* __restrict__ kh, u16* __restrict__ kl,
                u16* __restrict__ vh, u16* __restrict__ vl) {
  __shared__ __align__(16) unsigned int XsH[2][64][20], XsL[2][64][20];
  __shared__ __align__(16) unsigned int WsH[2][32][20], WsL[2][32][20];
  const int m0 = blockIdx.y * 64;
  const int n0 = blockIdx.x * 32;

  v8f acc = gemm_tile_planes(XH, XL, WH, WL, m0, n0, D_ / 2, XsH, XsL, WsH, WsL);

  const int lane = threadIdx.x & 31, wave = threadIdx.x >> 5;
  const int l15 = lane & 15, half = lane >> 4;
  const int nG = n0 + (wave & 1) * 16 + l15;      // 0..3071
  const float bias = b_qkv[nG];
  const int sec = nG >> 10;                        // 0=Q 1=K 2=V (uniform/tile)
  const int e   = nG & 1023;
  const int h   = e >> 6, hd = e & 63;

  #pragma unroll
  for (int v = 0; v < 8; ++v) {
    int m = m0 + (wave >> 1) * 16 + half * 8 + v;  // global token row
    int b = m >> 11, t = m & (T_ - 1);
    size_t bh = (size_t)b * H_ + h;
    if (sec == 0) {
      bf2 s = split_bf16((acc[v] + bias) * 0.125f);
      size_t idx = (bh * T_ + t) * HD_ + hd;
      qh[idx] = s.hi; ql[idx] = s.lo;
    } else if (sec == 1) {
      bf2 s = split_bf16(acc[v] + bias);
      size_t idx = (bh * T_ + t) * HD_ + hd;
      kh[idx] = s.hi; kl[idx] = s.lo;
    } else {
      bf2 s = split_bf16(acc[v] + bias);
      size_t idx = (bh * HD_ + hd) * T_ + t;       // transposed
      vh[idx] = s.hi; vl[idx] = s.lo;
    }
  }
}

// ---------------------------------------------------------------------------
// Kernel 2: causal flash attention on pre-split planes.
// Block = (b,h, 128 q rows); 8 waves each own a 16-row q tile.
// Double-buffered async K/V staging; P transpose stays per-wave (no barrier:
// a wave's LDS ops execute in order, ISA 7.3).
// ---------------------------------------------------------------------------
__global__ __launch_bounds__(256)
void attn_kernel(const unsigned int* __restrict__ QH, const unsigned int* __restrict__ QL,
                 const unsigned int* __restrict__ KH, const unsigned int* __restrict__ KL,
                 const unsigned int* __restrict__ VH, const unsigned int* __restrict__ VL,
                 u16* __restrict__ yh, u16* __restrict__ yl) {
  __shared__ __align__(16) unsigned int KsH[2][32][36], KsL[2][32][36]; // [krow][hd pairs]
  __shared__ __align__(16) unsigned int VsH[2][64][20], VsL[2][64][20]; // [hd][krow pairs]
  __shared__ __align__(16) u16 PsH[8][16][40], PsL[8][16][40];          // 20-word rows

  const int tid = threadIdx.x, lane = tid & 31, wave = tid >> 5;
  const int l15 = lane & 15, half = lane >> 4;
  const int bh = blockIdx.y;                 // b*H + h
  const int q0 = blockIdx.x * 128;
  const int kr = tid >> 3, kc = (tid & 7) * 4;   // K stage: 1 uint4/thread/plane
  const int vr = tid >> 2, vc = (tid & 3) * 4;   // V stage: 1 uint4/thread/plane

  auto stage = [&](int buf, int kbi) {
    size_t gk = (((size_t)bh * T_ + kbi * 32 + kr) * HD_ >> 1) + kc;
    async_b128(&KsH[buf][kr][kc], KH + gk);
    async_b128(&KsL[buf][kr][kc], KL + gk);
    size_t gv = (((size_t)bh * HD_ + vr) * T_ >> 1) + kbi * 16 + vc;
    async_b128(&VsH[buf][vr][vc], VH + gv);
    async_b128(&VsL[buf][vr][vc], VL + gv);
  };

  // Q fragments (16 rows x 64, scale folded): two B128 global loads per plane
  v16bf qhi[2], qlo[2];
  {
    const size_t rowbase = (((size_t)bh * T_ + q0 + wave * 16 + l15) * HD_) >> 1;
    #pragma unroll
    for (int ch = 0; ch < 2; ++ch) {
      qhi[ch] = frag_row(QH + rowbase, half, ch * 16);
      qlo[ch] = frag_row(QL + rowbase, half, ch * 16);
    }
  }

  v8f o0 = {}, o1 = {}, o2 = {}, o3 = {};
  float rmax[8], rsum[8];
  #pragma unroll
  for (int v = 0; v < 8; ++v) { rmax[v] = -__builtin_inff(); rsum[v] = 0.f; }

  const int nkb = blockIdx.x * 4 + 4;        // causal: k rows < q0+128
  stage(0, 0);                               // prologue
  int buf = 0;
  for (int kbi = 0; kbi < nkb; ++kbi, buf ^= 1) {
    wait_async();                            // own tile-t DMAs done
    __syncthreads();                         // everyone's done
    if (kbi + 1 < nkb) stage(buf ^ 1, kbi + 1);   // overlap next DMA w/ math

    // S = Qs @ K^T : two 16x16 tiles, K-dim 64 = 2 chunks of 32
    v8f s0 = {}, s1 = {};
    #pragma unroll
    for (int nt = 0; nt < 2; ++nt) {
      v8f sc = {};
      const int n = nt * 16 + l15;                // k-row within the 32-block
      #pragma unroll
      for (int ch = 0; ch < 2; ++ch) {
        v16bf bh_ = frag_row(&KsH[buf][n][0], half, ch * 16);
        v16bf bl_ = frag_row(&KsL[buf][n][0], half, ch * 16);
        sc = wmma3(qhi[ch], qlo[ch], bh_, bl_, sc);
      }
      if (nt == 0) s0 = sc; else s1 = sc;
    }

    // causal mask (C layout: lane -> column, VGPR v -> row v + 8*half)
    const int k0c = kbi * 32;
    #pragma unroll
    for (int v = 0; v < 8; ++v) {
      int row = q0 + wave * 16 + half * 8 + v;
      if (k0c + l15 > row)      s0[v] = -__builtin_inff();
      if (k0c + 16 + l15 > row) s1[v] = -__builtin_inff();
    }

    // online softmax (per-row stats replicated across each 16-lane half)
    float alpha[8], p0[8], p1[8];
    #pragma unroll
    for (int v = 0; v < 8; ++v) {
      float tm = rmax16(fmaxf(s0[v], s1[v]));
      float nm = fmaxf(rmax[v], tm);
      float a  = __expf(rmax[v] - nm);            // exp(-inf)=0 on first block
      float e0 = __expf(s0[v] - nm);
      float e1 = __expf(s1[v] - nm);
      rsum[v] = rsum[v] * a + rsum16(e0 + e1);
      rmax[v] = nm;
      alpha[v] = a; p0[v] = e0; p1[v] = e1;
    }
    #pragma unroll
    for (int v = 0; v < 8; ++v) {
      o0[v] *= alpha[v]; o1[v] *= alpha[v]; o2[v] *= alpha[v]; o3[v] *= alpha[v];
    }

    // P: split hi/lo, transpose C layout -> row-major bf16 in per-wave LDS.
    // Same-wave LDS ops are executed in order -> no block barrier needed.
    #pragma unroll
    for (int v = 0; v < 8; ++v) {
      int rl = half * 8 + v;
      bf2 a = split_bf16(p0[v]);
      bf2 b = split_bf16(p1[v]);
      PsH[wave][rl][l15] = a.hi;      PsL[wave][rl][l15] = a.lo;
      PsH[wave][rl][16 + l15] = b.hi; PsL[wave][rl][16 + l15] = b.lo;
    }

    v16bf phi = frag_row((const unsigned int*)&PsH[wave][l15][0], half, 0);
    v16bf plo = frag_row((const unsigned int*)&PsL[wave][l15][0], half, 0);

    // O += P @ V : 4 N-tiles of 16 (hd = 64), K-dim 32
    #pragma unroll
    for (int j = 0; j < 4; ++j) {
      const int n = j * 16 + l15;                 // hd column (row of Vt)
      v16bf vh_ = frag_row(&VsH[buf][n][0], half, 0);
      v16bf vl_ = frag_row(&VsL[buf][n][0], half, 0);
      if (j == 0)      o0 = wmma3(phi, plo, vh_, vl_, o0);
      else if (j == 1) o1 = wmma3(phi, plo, vh_, vl_, o1);
      else if (j == 2) o2 = wmma3(phi, plo, vh_, vl_, o2);
      else             o3 = wmma3(phi, plo, vh_, vl_, o3);
    }
  }

  // normalize, split, write y planes in [B,T,D] layout (d = h*64 + hd)
  const int b = bh >> 4, h = bh & 15;
  #pragma unroll
  for (int v = 0; v < 8; ++v) {
    int t = q0 + wave * 16 + half * 8 + v;
    float inv = 1.f / rsum[v];
    size_t base = ((size_t)b * T_ + t) * D_ + h * HD_;
    bf2 s;
    s = split_bf16(o0[v] * inv); yh[base +  0 + l15] = s.hi; yl[base +  0 + l15] = s.lo;
    s = split_bf16(o1[v] * inv); yh[base + 16 + l15] = s.hi; yl[base + 16 + l15] = s.lo;
    s = split_bf16(o2[v] * inv); yh[base + 32 + l15] = s.hi; yl[base + 32 + l15] = s.lo;
    s = split_bf16(o3[v] * inv); yh[base + 48 + l15] = s.hi; yl[base + 48 + l15] = s.lo;
  }
}

// ---------------------------------------------------------------------------
// Kernel 3: output projection  out = y @ w_proj^T + b_proj  (fp32 out)
// ---------------------------------------------------------------------------
__global__ __launch_bounds__(256)
void proj_kernel(const unsigned int* __restrict__ YH, const unsigned int* __restrict__ YL,
                 const unsigned int* __restrict__ WH, const unsigned int* __restrict__ WL,
                 const float* __restrict__ b_proj, float* __restrict__ out) {
  __shared__ __align__(16) unsigned int XsH[2][64][20], XsL[2][64][20];
  __shared__ __align__(16) unsigned int WsH[2][32][20], WsL[2][32][20];
  const int m0 = blockIdx.y * 64;
  const int n0 = blockIdx.x * 32;

  v8f acc = gemm_tile_planes(YH, YL, WH, WL, m0, n0, D_ / 2, XsH, XsL, WsH, WsL);

  const int lane = threadIdx.x & 31, wave = threadIdx.x >> 5;
  const int l15 = lane & 15, half = lane >> 4;
  const int nG = n0 + (wave & 1) * 16 + l15;
  const float bias = b_proj[nG];
  #pragma unroll
  for (int v = 0; v < 8; ++v) {
    int m = m0 + (wave >> 1) * 16 + half * 8 + v;
    out[(size_t)m * D_ + nG] = acc[v] + bias;
  }
}

// ---------------------------------------------------------------------------
extern "C" void kernel_launch(void* const* d_in, const int* in_sizes, int n_in,
                              void* d_out, int out_size, void* d_ws, size_t ws_size,
                              hipStream_t stream) {
  const float* x      = (const float*)d_in[0];
  // d_in[1] = attn_mask (bool causal tril) -- applied analytically in-kernel
  const float* w_qkv  = (const float*)d_in[2];
  const float* b_qkv  = (const float*)d_in[3];
  const float* w_proj = (const float*)d_in[4];
  const float* b_proj = (const float*)d_in[5];
  float* out = (float*)d_out;

  const size_t NX  = (size_t)M_ * D_;        // 4,194,304
  const size_t NWQ = (size_t)3 * D_ * D_;    // 3,145,728
  const size_t NWP = (size_t)D_ * D_;        // 1,048,576

  u16* p = (u16*)d_ws;
  u16* xh  = p; p += NX;   u16* xl  = p; p += NX;
  u16* wqh = p; p += NWQ;  u16* wql = p; p += NWQ;
  u16* wph = p; p += NWP;  u16* wpl = p; p += NWP;
  u16* qh  = p; p += NX;   u16* ql  = p; p += NX;
  u16* kh  = p; p += NX;   u16* kl  = p; p += NX;
  u16* vh  = p; p += NX;   u16* vl  = p; p += NX;
  u16* yh  = p; p += NX;   u16* yl  = p; p += NX;

  split_kernel<<<dim3((unsigned)((NX  + 255) / 256)), 256, 0, stream>>>(x,      xh,  xl,  (int)NX);
  split_kernel<<<dim3((unsigned)((NWQ + 255) / 256)), 256, 0, stream>>>(w_qkv,  wqh, wql, (int)NWQ);
  split_kernel<<<dim3((unsigned)((NWP + 255) / 256)), 256, 0, stream>>>(w_proj, wph, wpl, (int)NWP);

  qkv_kernel<<<dim3(3 * D_ / 32, M_ / 64), 256, 0, stream>>>(
      (const unsigned int*)xh, (const unsigned int*)xl,
      (const unsigned int*)wqh, (const unsigned int*)wql,
      b_qkv, qh, ql, kh, kl, vh, vl);

  attn_kernel<<<dim3(T_ / 128, B_ * H_), 256, 0, stream>>>(
      (const unsigned int*)qh, (const unsigned int*)ql,
      (const unsigned int*)kh, (const unsigned int*)kl,
      (const unsigned int*)vh, (const unsigned int*)vl,
      yh, yl);

  proj_kernel<<<dim3(D_ / 32, M_ / 64), 256, 0, stream>>>(
      (const unsigned int*)yh, (const unsigned int*)yl,
      (const unsigned int*)wph, (const unsigned int*)wpl,
      b_proj, out);
}
